// GlobalCellGraph_40793599377596
// MI455X (gfx1250) — matrix operand order repaired and checked by reading, not compiled, not run on hardware
//
#include <hip/hip_runtime.h>
#include <hip/hip_bf16.h>
#include <math.h>

// GlobalCellGraph on MI455X (gfx1250, wave32, WMMA).
// N=6144 cells, D=128, H=4 heads, DK=32.
//
// Algebraic fusion: s = (query@Wq + bq) * wo_perhead / sqrt(32)  @  (key@Wk + bk)^T
// => one N x N x 128 GEMM. bo cancels under softmax. adj[n,m] = s[n,m] >= t[n],
// t[n] = rowmax + log(phi) + log(rowsumexp). Never materialize the 151MB score
// matrix: pass 1 computes t[n] (online logsumexp over WMMA tiles), pass 2
// recomputes tiles (operands are L2-resident, 3MB total) and thresholds.

#define N_CELLS 6144
#define DIM     128

typedef __attribute__((ext_vector_type(16))) _Float16 v16h;
typedef __attribute__((ext_vector_type(8)))  _Float16 h8;
typedef __attribute__((ext_vector_type(8)))  float    v8f;

// ---- WMMA fragment loaders (per ISA 7.12.2 VGPR layouts, wave32) ----------

// A matrix 16x32 f16: lane L holds row (L&15); VGPRs 0..3 hold K = kb..kb+7,
// VGPRs 4..7 hold K = 16+kb..16+kb+7, kb = (L<16) ? 0 : 8. Two 16B chunks.
__device__ __forceinline__ v16h load_a_frag(const _Float16* __restrict__ qrows,
                                            int lane, int kc) {
  const int r  = lane & 15;
  const int kb = (lane < 16) ? 0 : 8;
  const _Float16* p = qrows + r * DIM + kc * 32 + kb;
  union { v16h v; h8 h[2]; } u;
  u.h[0] = *(const h8*)(p);
  u.h[1] = *(const h8*)(p + 16);
  return u.v;
}

// B matrix 32x16 f16: lane L holds column (L&15); VGPRs 0..3 hold
// K = kb..kb+7, VGPRs 4..7 hold K = 16+kb..16+kb+7, kb = (L<16) ? 0 : 8.
// B[k][col] = keff[col][k] (K^T), so two 16B chunks of keff row `col`.
__device__ __forceinline__ v16h load_b_frag(const _Float16* __restrict__ keff,
                                            int colbase, int lane, int kc) {
  const int col = colbase + (lane & 15);
  const int kb  = (lane < 16) ? 0 : 8;
  const _Float16* p = keff + (size_t)col * DIM + kc * 32 + kb;
  union { v16h v; h8 h[2]; } u;
  u.h[0] = *(const h8*)(p);
  u.h[1] = *(const h8*)(p + 16);
  return u.v;
}

// ---- Pass 0: fused projections -> f16 qeff (wo/sqrt(DK) folded in), keff ---

__global__ void __launch_bounds__(128)
gcg_proj_kernel(const float* __restrict__ query, const float* __restrict__ key_feat,
                const float* __restrict__ Wq, const float* __restrict__ bq,
                const float* __restrict__ Wk, const float* __restrict__ bk,
                const float* __restrict__ wo,
                _Float16* __restrict__ qeff, _Float16* __restrict__ keff) {
  __shared__ float row[DIM];
  const int n = blockIdx.x;
  const int c = threadIdx.x;

  row[c] = query[(size_t)n * DIM + c];
  __syncthreads();
  float acc = 0.f;
  #pragma unroll 8
  for (int d = 0; d < DIM; ++d) acc = fmaf(row[d], Wq[d * DIM + c], acc);
  acc += bq[c];
  // fold head-combine weight and 1/sqrt(32) into q
  qeff[(size_t)n * DIM + c] =
      (_Float16)(acc * wo[c >> 5] * 0.17677669529663687f);
  __syncthreads();

  row[c] = key_feat[(size_t)n * DIM + c];
  __syncthreads();
  acc = 0.f;
  #pragma unroll 8
  for (int d = 0; d < DIM; ++d) acc = fmaf(row[d], Wk[d * DIM + c], acc);
  acc += bk[c];
  keff[(size_t)n * DIM + c] = (_Float16)acc;
}

// ---- Pass 1: per-row threshold t[n] via online logsumexp over WMMA tiles ---
// One wave owns 16 rows; A fragments (16x128) held in registers for the whole
// sweep over all 384 column tiles. D layout: VGPR v <-> row v (lanes 0-15) or
// row v+8 (lanes 16-31); lane&15 <-> column.

__global__ void __launch_bounds__(256)
gcg_stats_kernel(const _Float16* __restrict__ qeff, const _Float16* __restrict__ keff,
                 const float* __restrict__ phi_p, float* __restrict__ trow) {
  const int lane = threadIdx.x & 31;
  const int wid  = blockIdx.x * (blockDim.x >> 5) + (threadIdx.x >> 5);
  const int rowbase = wid * 16;

  v16h a[4];
  #pragma unroll
  for (int kc = 0; kc < 4; ++kc)
    a[kc] = load_a_frag(qeff + (size_t)rowbase * DIM, lane, kc);

  float m[8], s[8];
  #pragma unroll
  for (int v = 0; v < 8; ++v) { m[v] = -__builtin_inff(); s[v] = 0.f; }

  for (int ct = 0; ct < N_CELLS / 16; ++ct) {
    v8f c = {};
    #pragma unroll
    for (int kc = 0; kc < 4; ++kc) {
      v16h b = load_b_frag(keff, ct * 16, lane, kc);
      c = __builtin_amdgcn_wmma_f32_16x16x32_f16(
              false, a[kc], false, b, (short)0, c, false, false);
    }
    #pragma unroll
    for (int v = 0; v < 8; ++v) {     // online max / sumexp per row-stream
      const float x  = c[v];
      const float nm = fmaxf(m[v], x);
      s[v] = s[v] * __expf(m[v] - nm) + __expf(x - nm);
      m[v] = nm;
    }
  }

  // merge the 16 column-streams of each row (shuffle tree within half-wave)
  #pragma unroll
  for (int v = 0; v < 8; ++v) {
    float mv = m[v], sv = s[v];
    #pragma unroll
    for (int off = 1; off < 16; off <<= 1) {
      const float om = __shfl_xor(mv, off, 16);
      const float os = __shfl_xor(sv, off, 16);
      const float nm = fmaxf(mv, om);
      sv = sv * __expf(mv - nm) + os * __expf(om - nm);
      mv = nm;
    }
    m[v] = mv; s[v] = sv;
  }

  if ((lane & 15) == 0) {
    const float lphi = __logf(*phi_p);
    const int rb = rowbase + ((lane < 16) ? 0 : 8);
    #pragma unroll
    for (int v = 0; v < 8; ++v)
      trow[rb + v] = m[v] + __logf(s[v]) + lphi;
  }
}

// ---- Pass 2: recompute tiles, threshold, store binary adjacency ------------
// One wave per 16x64 output strip: A reused across 4 column tiles, 16 WMMAs.

__global__ void __launch_bounds__(256)
gcg_adj_kernel(const _Float16* __restrict__ qeff, const _Float16* __restrict__ keff,
               const float* __restrict__ trow, float* __restrict__ adj) {
  const int lane = threadIdx.x & 31;
  const int wid  = blockIdx.x * (blockDim.x >> 5) + (threadIdx.x >> 5);
  const int rowgroup = wid / (N_CELLS / 64);
  const int colstrip = wid % (N_CELLS / 64);
  const int rowbase  = rowgroup * 16;

  v16h a[4];
  #pragma unroll
  for (int kc = 0; kc < 4; ++kc)
    a[kc] = load_a_frag(qeff + (size_t)rowbase * DIM, lane, kc);

  const int rbk = rowbase + ((lane < 16) ? 0 : 8);
  float tv[8];
  #pragma unroll
  for (int v = 0; v < 8; ++v) tv[v] = trow[rbk + v];

  #pragma unroll
  for (int sub = 0; sub < 4; ++sub) {
    const int colbase = colstrip * 64 + sub * 16;
    v8f c = {};
    #pragma unroll
    for (int kc = 0; kc < 4; ++kc) {
      v16h b = load_b_frag(keff, colbase, lane, kc);
      c = __builtin_amdgcn_wmma_f32_16x16x32_f16(
              false, a[kc], false, b, (short)0, c, false, false);
    }
    const int col = colbase + (lane & 15);
    #pragma unroll
    for (int v = 0; v < 8; ++v)
      adj[(size_t)(rbk + v) * N_CELLS + col] = (c[v] >= tv[v]) ? 1.0f : 0.0f;
  }
}

// ---- launch ---------------------------------------------------------------

extern "C" void kernel_launch(void* const* d_in, const int* in_sizes, int n_in,
                              void* d_out, int out_size, void* d_ws, size_t ws_size,
                              hipStream_t stream) {
  const float* query    = (const float*)d_in[0];
  const float* key_feat = (const float*)d_in[1];
  const float* Wq       = (const float*)d_in[2];
  const float* bq       = (const float*)d_in[3];
  const float* Wk       = (const float*)d_in[4];
  const float* bk       = (const float*)d_in[5];
  const float* wo       = (const float*)d_in[6];
  // d_in[7] = bo: provably cancels (softmax shift invariance) -> unused
  const float* phi      = (const float*)d_in[8];

  // scratch layout (~3.2 MB): qeff f16 | keff f16 | t[n] f32
  char* ws = (char*)d_ws;
  _Float16* qeff = (_Float16*)ws;
  _Float16* keff = qeff + (size_t)N_CELLS * DIM;
  float*    trow = (float*)(ws + 2 * (size_t)N_CELLS * DIM * sizeof(_Float16));

  float* adj = (float*)d_out;

  gcg_proj_kernel<<<N_CELLS, DIM, 0, stream>>>(
      query, key_feat, Wq, bq, Wk, bk, wo, qeff, keff);

  // 384 row-groups of 16 rows, 8 waves per block -> 48 blocks
  gcg_stats_kernel<<<(N_CELLS / 16) / 8, 256, 0, stream>>>(qeff, keff, phi, trow);

  // 384 row-groups x 96 col-strips = 36864 waves, 8 per block -> 4608 blocks
  gcg_adj_kernel<<<(N_CELLS / 16) * (N_CELLS / 64) / 8, 256, 0, stream>>>(
      qeff, keff, trow, adj);
}
